// ScaledDotProductAttention_3753801416775
// MI455X (gfx1250) — matrix operand order repaired and checked by reading, not compiled, not run on hardware
//
#include <hip/hip_runtime.h>

typedef __attribute__((ext_vector_type(16))) __bf16 v16bf;
typedef __attribute__((ext_vector_type(8)))  float  v8f;
typedef __attribute__((ext_vector_type(4)))  int    v4i;

#define Bn 4
#define Hn 16
#define Sn 2048
#define Dn 64
#define WAVES 4
#define SCALE 0.125f   /* 1/sqrt(64) */

// ---- CDNA5 async-copy-to-LDS plumbing (builtin preferred, inline asm fallback)
#if defined(__AMDGCN__) && __has_builtin(__builtin_amdgcn_global_load_async_to_lds_b128)
#define HAVE_ASYNC_BUILTIN 1
#endif
#if defined(__AMDGCN__) && __has_builtin(__builtin_amdgcn_s_wait_asynccnt)
#define WAIT_ASYNC(n) __builtin_amdgcn_s_wait_asynccnt(n)
#else
#define WAIT_ASYNC(n) asm volatile("s_wait_asynccnt " #n ::: "memory")
#endif

// Copy 16 bytes global -> LDS through the async path (ASYNCcnt-tracked,
// no VGPR round trip). One chunk per lane per call.
__device__ __forceinline__ void async_cp16(const float* g, float* l)
{
#ifdef HAVE_ASYNC_BUILTIN
    // Builtin signature (probed from hipcc diagnostics): (int4*, int4*, Ii, Ii)
    __builtin_amdgcn_global_load_async_to_lds_b128((v4i*)g, (v4i*)l, 0, 0);
#else
    // Generic LDS pointer: low 32 bits are the LDS byte address.
    unsigned loff = (unsigned)(unsigned long long)l;
    unsigned long long ga = (unsigned long long)g;
    asm volatile("global_load_async_to_lds_b128 %0, %1, off"
                 :: "v"(loff), "v"(ga) : "memory");
#endif
}

// Fused scaled-dot-product attention producing (out, weights).
// One wave = 16 query rows; 4 waves share one (b,h) 64-query block and share
// double-buffered K/V tiles staged to LDS with async copies.
// Pass A: online row max/sum over all keys. Pass B: recompute scores, write
// normalized weights (NT streaming stores), accumulate out = W @ V via WMMA.
__global__ __launch_bounds__(WAVES * 32) void sdpa_fused(
    const float* __restrict__ Q, const float* __restrict__ K,
    const float* __restrict__ V, const int* __restrict__ Mk,
    float* __restrict__ Out, float* __restrict__ Wout)
{
    __shared__ float  Kb[2][32][68];          // double-buffered K tile (padded)
    __shared__ float  Vb[2][32][68];          // double-buffered V tile (padded)
    __shared__ __bf16 lds_w[WAVES][16][34];   // per-wave W-transpose buffer

    const int tid   = threadIdx.x;
    const int lane  = tid & 31;
    const int wave  = tid >> 5;
    const int wgq   = blockIdx.x & 31;          // 32 query blocks of 64
    const int bh    = blockIdx.x >> 5;          // (b*H + h)
    const int b     = bh >> 4;                  // H == 16
    const int qbase = wgq * 64 + wave * 16;     // this wave's 16 query rows

    const int half   = lane >> 4;               // lane half (0/1)
    const int l15    = lane & 15;
    const int kbaseA = half * 8;                // unused; A loads use K offset half * 16

    const float* Qp = Q   + (size_t)bh * Sn * Dn;
    const float* Kp = K   + (size_t)bh * Sn * Dn;
    const float* Vp = V   + (size_t)bh * Sn * Dn;
    const int*   Mp = Mk  + (size_t)b  * Sn * Sn;   // mask shared over heads
    float*       Op = Out + (size_t)bh * Sn * Dn;
    float*       Wp = Wout+ (size_t)bh * Sn * Sn;

    // ---- Load Q A-fragments once (two K=32 chunks covering D=64) ----
    v16bf aq[2];
    {
        const float* qrow = Qp + (size_t)(qbase + l15) * Dn;
        #pragma unroll
        for (int kh = 0; kh < 2; ++kh) {
            #pragma unroll
            for (int e = 0; e < 8; ++e) {
                aq[kh][e]     = (__bf16)qrow[kh * 32 + half * 16 + e];
                aq[kh][e + 8] = (__bf16)qrow[kh * 32 + half * 16 + 8 + e];
            }
        }
    }

    // ---- Pass A: online softmax statistics ----
    float pm[8], ps[8];
    #pragma unroll
    for (int r = 0; r < 8; ++r) { pm[r] = -__builtin_inff(); ps[r] = 0.0f; }

    // Prologue: stage K tile 0 (16 rows x 64 f32 = 256 x 16B chunks, 2/thread).
    #pragma unroll
    for (int i = 0; i < 2; ++i) {
        const int c = tid + i * 128, row = c >> 4, col = (c & 15) * 4;
        async_cp16(Kp + (size_t)row * Dn + col, &Kb[0][row][col]);
    }

    for (int kt = 0; kt < Sn / 16; ++kt) {
        const int cur = kt & 1;
        if (kt + 1 < Sn / 16) {       // prefetch next tile into other buffer
            const int kbn = (kt + 1) * 16;
            #pragma unroll
            for (int i = 0; i < 2; ++i) {
                const int c = tid + i * 128, row = c >> 4, col = (c & 15) * 4;
                async_cp16(Kp + (size_t)(kbn + row) * Dn + col,
                           &Kb[cur ^ 1][row][col]);
            }
            WAIT_ASYNC(2);            // current tile's copies have landed
        } else {
            WAIT_ASYNC(0);
        }
        __syncthreads();              // tile visible to all waves

        v8f c = {};
        #pragma unroll
        for (int kh = 0; kh < 2; ++kh) {
            const float* krow = &Kb[cur][l15][kh * 32 + half * 16];
            v16bf bk;
            #pragma unroll
            for (int e = 0; e < 16; ++e) bk[e] = (__bf16)krow[e];
            c = __builtin_amdgcn_wmma_f32_16x16x32_bf16(
                    false, aq[kh], false, bk, (short)0, c, false, false);
        }
        const int key = kt * 16 + l15;
        #pragma unroll
        for (int r = 0; r < 8; ++r) {
            const int qrow = qbase + r + half * 8;
            const float sc = c[r] * SCALE;
            const int  mk  = Mp[(size_t)qrow * Sn + key];
            const float m  = mk ? sc : -1e9f;
            const float nm = fmaxf(pm[r], m);
            ps[r] = ps[r] * __expf(pm[r] - nm) + __expf(m - nm);
            pm[r] = nm;
        }
        __syncthreads();              // done reading before buffer is reused
    }

    // Reduce row stats across the 16 lanes of each half.
    #pragma unroll
    for (int off = 1; off < 16; off <<= 1) {
        #pragma unroll
        for (int r = 0; r < 8; ++r) {
            const float om = __shfl_xor(pm[r], off, 32);
            const float os = __shfl_xor(ps[r], off, 32);
            const float nm = fmaxf(pm[r], om);
            ps[r] = ps[r] * __expf(pm[r] - nm) + os * __expf(om - nm);
            pm[r] = nm;
        }
    }
    float rsinv[8];
    #pragma unroll
    for (int r = 0; r < 8; ++r) rsinv[r] = 1.0f / ps[r];

    // ---- Pass B: weights out + O = W @ V ----
    v8f oacc[4] = {};

    // Prologue: stage K+V chunk 0 (32 rows each; 4+4 chunks per thread).
    #pragma unroll
    for (int i = 0; i < 4; ++i) {
        const int c = tid + i * 128, row = c >> 4, col = (c & 15) * 4;
        async_cp16(Kp + (size_t)row * Dn + col, &Kb[0][row][col]);
        async_cp16(Vp + (size_t)row * Dn + col, &Vb[0][row][col]);
    }

    for (int kc = 0; kc < Sn / 32; ++kc) {
        const int cur = kc & 1;
        const int kb2 = kc * 32;
        if (kc + 1 < Sn / 32) {
            const int nb = (kc + 1) * 32;
            #pragma unroll
            for (int i = 0; i < 4; ++i) {
                const int c = tid + i * 128, row = c >> 4, col = (c & 15) * 4;
                async_cp16(Kp + (size_t)(nb + row) * Dn + col,
                           &Kb[cur ^ 1][row][col]);
                async_cp16(Vp + (size_t)(nb + row) * Dn + col,
                           &Vb[cur ^ 1][row][col]);
            }
            WAIT_ASYNC(8);
        } else {
            WAIT_ASYNC(0);
        }
        __syncthreads();

        #pragma unroll
        for (int sub = 0; sub < 2; ++sub) {
            v8f c = {};
            #pragma unroll
            for (int kh = 0; kh < 2; ++kh) {
                const float* krow = &Kb[cur][sub * 16 + l15][kh * 32 + half * 16];
                v16bf bk;
                #pragma unroll
                for (int e = 0; e < 16; ++e) bk[e] = (__bf16)krow[e];
                c = __builtin_amdgcn_wmma_f32_16x16x32_bf16(
                        false, aq[kh], false, bk, (short)0, c, false, false);
            }
            const int key = kb2 + sub * 16 + l15;
            #pragma unroll
            for (int r = 0; r < 8; ++r) {
                const int M    = r + half * 8;
                const int qrow = qbase + M;
                const float sc = c[r] * SCALE;
                const int  mk  = Mp[(size_t)qrow * Sn + key];
                const float m  = mk ? sc : -1e9f;
                const float w  = __expf(m - pm[r]) * rsinv[r];
                // 1 GB write-once stream: keep it out of the caches
                __builtin_nontemporal_store(w, &Wp[(size_t)qrow * Sn + key]);
                lds_w[wave][M][sub * 16 + l15] = (__bf16)w;
            }
        }
        // Same-wave LDS ops are in-order; just stop compiler reordering.
        __builtin_amdgcn_wave_barrier();

        // Re-read W tile as an A-fragment (16 q-rows x 32 keys).
        v16bf aw;
        #pragma unroll
        for (int e = 0; e < 8; ++e) {
            aw[e]     = lds_w[wave][l15][half * 16 + e];
            aw[e + 8] = lds_w[wave][l15][half * 16 + 8 + e];
        }
        __builtin_amdgcn_wave_barrier();

        // O(16x64) += W(16x32) @ V(32x64): 4 WMMAs over D tiles of 16.
        #pragma unroll
        for (int dt = 0; dt < 4; ++dt) {
            v16bf bv;
            #pragma unroll
            for (int e = 0; e < 16; ++e)
                bv[e] = (__bf16)Vb[cur][half * 16 + e][dt * 16 + l15];
            oacc[dt] = __builtin_amdgcn_wmma_f32_16x16x32_bf16(
                           false, aw, false, bv, (short)0, oacc[dt], false, false);
        }
        __syncthreads();              // all waves done with this buffer
    }

    // ---- Write out[16 x 64] ----
    #pragma unroll
    for (int dt = 0; dt < 4; ++dt) {
        #pragma unroll
        for (int r = 0; r < 8; ++r) {
            const int qrow = qbase + r + half * 8;
            __builtin_nontemporal_store(oacc[dt][r],
                                        &Op[(size_t)qrow * Dn + dt * 16 + l15]);
        }
    }
}

extern "C" void kernel_launch(void* const* d_in, const int* in_sizes, int n_in,
                              void* d_out, int out_size, void* d_ws, size_t ws_size,
                              hipStream_t stream) {
    const float* Q  = (const float*)d_in[0];
    const float* K  = (const float*)d_in[1];
    const float* V  = (const float*)d_in[2];
    const int*   Mk = (const int*)d_in[3];
    float* out  = (float*)d_out;                       // (B,H,S,D)
    float* wout = out + (size_t)Bn * Hn * Sn * Dn;     // (B,H,S,S) follows
    dim3 grid(Bn * Hn * (Sn / 64));                    // 2048 workgroups
    sdpa_fused<<<grid, WAVES * 32, 0, stream>>>(Q, K, V, Mk, out, wout);
}
